// AttentionOp_27719718928719
// MI455X (gfx1250) — matrix operand order repaired and checked by reading, not compiled, not run on hardware
//
#include <hip/hip_runtime.h>
#include <hip/hip_bf16.h>
#include <stdint.h>

// GQA causal flash-attention for gfx1250 (MI455X), bf16 WMMA + f32 accumulation.
// Shapes fixed by reference: b=2, t=s=1024, n_q=32, n_kv=8, d=128.
// kv tile = 64, q tile = 16 rows/wave, 4 waves/block (64 q rows, one q head).

typedef __attribute__((ext_vector_type(16))) __bf16 v16bf;
typedef __attribute__((ext_vector_type(8)))  float  v8f;

#define TQ   1024
#define SKV  1024
#define NQ   32
#define NKV  8
#define DH   128
#define KVT  64                      // kv tile width
#define MASKVAL (-2.3819763e38f)     // -0.7 * FLT_MAX

union Frag {
  uint32_t u[8];
  v16bf    v;
};

__device__ __forceinline__ uint32_t pack_bf16x2(float a, float b) {
  unsigned short ha = __builtin_bit_cast(unsigned short, (__bf16)a);
  unsigned short hb = __builtin_bit_cast(unsigned short, (__bf16)b);
  return (uint32_t)ha | ((uint32_t)hb << 16);
}

__device__ __forceinline__ float grp16_max(float x) {
  x = fmaxf(x, __shfl_xor(x, 1, 32));
  x = fmaxf(x, __shfl_xor(x, 2, 32));
  x = fmaxf(x, __shfl_xor(x, 4, 32));
  x = fmaxf(x, __shfl_xor(x, 8, 32));
  return x;
}
__device__ __forceinline__ float grp16_sum(float x) {
  x += __shfl_xor(x, 1, 32);
  x += __shfl_xor(x, 2, 32);
  x += __shfl_xor(x, 4, 32);
  x += __shfl_xor(x, 8, 32);
  return x;
}

__global__ __launch_bounds__(128) void flash_gqa_kernel(
    const float* __restrict__ Q, const float* __restrict__ K,
    const float* __restrict__ V, const int* __restrict__ SEG,
    float* __restrict__ OUT) {
  __shared__ __bf16 ldsK[KVT * DH];    // K tile   [kv][d]        16 KB
  __shared__ __bf16 ldsVt[DH * KVT];   // V tile   [d][kv]        16 KB
  __shared__ __bf16 ldsP[4][16 * KVT]; // per-wave P [row][kv]     8 KB
  __shared__ int ldsSegK[KVT];
  __shared__ int ldsSegQ[64];

  const int tid  = threadIdx.x;
  const int wave = tid >> 5;
  const int lane = tid & 31;
  const int l15  = lane & 15;
  const int hi   = lane >> 4;   // 0: lanes 0-15, 1: lanes 16-31

  const int qblk  = blockIdx.x & 15;          // 16 blocks of 64 q rows
  const int head  = (blockIdx.x >> 4) & 31;
  const int batch = blockIdx.x >> 9;
  const int kvh   = head >> 2;                // GQA group size 4
  const int wgQ   = qblk * 64;
  const int waveQ = wgQ + wave * 16;

  // ---- hoist Q A-fragments into registers (loop-invariant over kv) ----
  // A layout: M = lane&15 (both halves), element pair p holds
  // K = (p>>2)*16 + hi*8 + (p&3)*2 (+1)
  Frag qa[4];
  {
    const int qrow = waveQ + l15;
    const float* qb = Q + (((size_t)batch * TQ + qrow) * NQ + head) * DH;
#pragma unroll
    for (int kc = 0; kc < 4; ++kc) {
#pragma unroll
      for (int p = 0; p < 8; ++p) {
        int ks = ((p >> 2) << 4) + (hi << 3) + ((p & 3) << 1);
        float2 f = *(const float2*)(qb + kc * 32 + ks);
        qa[kc].u[p] = pack_bf16x2(f.x, f.y);
      }
    }
  }
  if (tid < 64) ldsSegQ[tid] = SEG[batch * SKV + wgQ + tid];

  // ---- per-row flash state (rows r and r+8 per lane group) ----
  float m[8], l[8];
  v8f accO[8];                 // accO[d-tile][row]
#pragma unroll
  for (int r = 0; r < 8; ++r) { m[r] = MASKVAL; l[r] = 0.0f; }
#pragma unroll
  for (int dt = 0; dt < 8; ++dt) { v8f z = {}; accO[dt] = z; }

  const int nKvTiles = qblk + 1;   // causal: kv <= wgQ + 63
  for (int kt = 0; kt < nKvTiles; ++kt) {
    const int kvbase = kt * KVT;
    __syncthreads();   // protect LDS K/V from previous iteration's readers

    // ---- stage K (row-major bf16) and V (transposed bf16), float4 loads ----
    for (int i = tid; i < (KVT * DH) / 4; i += 128) {
      int kv = i >> 5, d = (i & 31) << 2;
      size_t g = (((size_t)batch * SKV + kvbase + kv) * NKV + kvh) * DH + d;
      float4 kf = *(const float4*)(K + g);
      float4 vf = *(const float4*)(V + g);
      uint32_t* kd = (uint32_t*)&ldsK[kv * DH + d];
      kd[0] = pack_bf16x2(kf.x, kf.y);
      kd[1] = pack_bf16x2(kf.z, kf.w);
      ldsVt[(d + 0) * KVT + kv] = (__bf16)vf.x;
      ldsVt[(d + 1) * KVT + kv] = (__bf16)vf.y;
      ldsVt[(d + 2) * KVT + kv] = (__bf16)vf.z;
      ldsVt[(d + 3) * KVT + kv] = (__bf16)vf.w;
    }
    if (tid < KVT) ldsSegK[tid] = SEG[batch * SKV + kvbase + tid];
    __syncthreads();

    // ---- prefetch next kv tile while this one computes ----
    if (kt + 1 < nKvTiles) {
      int pkv = tid >> 1;
      size_t g = (((size_t)batch * SKV + kvbase + KVT + pkv) * NKV + kvh) * DH +
                 ((size_t)(tid & 1) << 6);
      __builtin_prefetch(K + g, 0, 0);
      __builtin_prefetch(V + g, 0, 0);
    }

    // ---- S = Q * K^T : four 16x16 f32 tiles over 4 d-chunks of 32 ----
    v8f c[4];
#pragma unroll
    for (int nt = 0; nt < 4; ++nt) { v8f z = {}; c[nt] = z; }
    const uint32_t* k32 = (const uint32_t*)&ldsK[0];
#pragma unroll
    for (int kc = 0; kc < 4; ++kc) {
#pragma unroll
      for (int nt = 0; nt < 4; ++nt) {
        Frag b;   // B layout: N = l15, K = (p>>2)*16 + hi*8 + (p&3)*2 (+1)
#pragma unroll
        for (int p = 0; p < 8; ++p) {
          int ks = ((p >> 2) << 4) + (hi << 3) + ((p & 3) << 1);
          b.u[p] = k32[((nt * 16 + l15) * DH + kc * 32 + ks) >> 1];
        }
        c[nt] = __builtin_amdgcn_wmma_f32_16x16x32_bf16(
            false, qa[kc].v, false, b.v, (short)0, c[nt], false, false);
      }
    }

    // ---- mask + online softmax; P -> wave-private LDS as bf16 ----
    int sk[4];
#pragma unroll
    for (int nt = 0; nt < 4; ++nt) sk[nt] = ldsSegK[nt * 16 + l15];
    __bf16* pw = &ldsP[wave][0];
#pragma unroll
    for (int r = 0; r < 8; ++r) {
      int row  = r + (hi << 3);
      int qrow = waveQ + row;
      int sq   = ldsSegQ[(wave << 4) + row];
      float s[4];
      bool ok[4];
      float tmax = MASKVAL;
#pragma unroll
      for (int nt = 0; nt < 4; ++nt) {
        ok[nt] = (kvbase + nt * 16 + l15 <= qrow) && (sk[nt] == sq);
        s[nt]  = ok[nt] ? c[nt][r] : MASKVAL;
        tmax   = fmaxf(tmax, s[nt]);
      }
      float mnew  = fmaxf(m[r], grp16_max(tmax));
      float scale = __expf(m[r] - mnew);
      float rs = 0.0f;
#pragma unroll
      for (int nt = 0; nt < 4; ++nt) {
        float pv = ok[nt] ? __expf(s[nt] - mnew) : 0.0f;
        rs += pv;
        pw[row * KVT + nt * 16 + l15] = (__bf16)pv;
      }
      l[r] = l[r] * scale + grp16_sum(rs);
      m[r] = mnew;
#pragma unroll
      for (int dt = 0; dt < 8; ++dt) accO[dt][r] *= scale;
    }

    // ---- O += P * V : 8 d-tiles x 2 k-chunks of 32 kv ----
    const uint32_t* p32 = (const uint32_t*)pw;
    const uint32_t* v32 = (const uint32_t*)&ldsVt[0];
    Frag pa[2];
#pragma unroll
    for (int kc = 0; kc < 2; ++kc) {
#pragma unroll
      for (int p = 0; p < 8; ++p) {
        int ks = ((p >> 2) << 4) + (hi << 3) + ((p & 3) << 1);
        pa[kc].u[p] = p32[(l15 * KVT + kc * 32 + ks) >> 1];
      }
    }
#pragma unroll
    for (int dt = 0; dt < 8; ++dt) {
#pragma unroll
      for (int kc = 0; kc < 2; ++kc) {
        Frag vb;
#pragma unroll
        for (int p = 0; p < 8; ++p) {
          int ks = ((p >> 2) << 4) + (hi << 3) + ((p & 3) << 1);
          vb.u[p] = v32[(((dt << 4) + l15) * KVT + kc * 32 + ks) >> 1];
        }
        accO[dt] = __builtin_amdgcn_wmma_f32_16x16x32_bf16(
            false, pa[kc].v, false, vb.v, (short)0, accO[dt], false, false);
      }
    }
  }

  // ---- normalize and store ----
#pragma unroll
  for (int r = 0; r < 8; ++r) {
    int row  = r + (hi << 3);
    int qrow = waveQ + row;
    float inv = 1.0f / l[r];
    size_t base = (((size_t)batch * TQ + qrow) * NQ + head) * DH;
#pragma unroll
    for (int dt = 0; dt < 8; ++dt) {
      OUT[base + (dt << 4) + l15] = accO[dt][r] * inv;
    }
  }
}

extern "C" void kernel_launch(void* const* d_in, const int* in_sizes, int n_in,
                              void* d_out, int out_size, void* d_ws, size_t ws_size,
                              hipStream_t stream) {
  (void)in_sizes; (void)n_in; (void)out_size; (void)d_ws; (void)ws_size;
  const float* Q   = (const float*)d_in[0];
  const float* K   = (const float*)d_in[1];
  const float* V   = (const float*)d_in[2];
  const int*   SEG = (const int*)d_in[3];
  float* OUT = (float*)d_out;
  // grid: batch(2) * heads(32) * q-blocks of 64 rows (16) = 1024 workgroups
  dim3 grid(2 * 32 * 16);
  flash_gqa_kernel<<<grid, 128, 0, stream>>>(Q, K, V, SEG, OUT);
}
